// GCN_60413009985908
// MI455X (gfx1250) — matrix operand order repaired and checked by reading, not compiled, not run on hardware
//
#include <hip/hip_runtime.h>
#include <cstddef>

// ---------------------------------------------------------------------------
// Problem constants (match reference)
// ---------------------------------------------------------------------------
#define N_NODES  20000
#define N_EDGES  160000
#define DIM      512
#define N_GRAPHS 128
#define MPAD     20096          // N_NODES rounded up to 128-row GEMM tiles

typedef __attribute__((ext_vector_type(16))) __bf16 v16bf;
typedef __attribute__((ext_vector_type(8)))  __bf16 bf16x8;
typedef __attribute__((ext_vector_type(8)))  float  v8f;

// CDNA5 async global->LDS copy, 16 bytes per active lane (ASYNCcnt-tracked).
__device__ __forceinline__ void async_cp16(unsigned lds_off, const void* gaddr) {
    asm volatile("global_load_async_to_lds_b128 %0, %1, off"
                 :: "v"(lds_off), "v"((unsigned long long)gaddr) : "memory");
}

// ---------------------------------------------------------------------------
// Small utility kernels
// ---------------------------------------------------------------------------
__global__ void k_fill_f32(float* p, int n, float v) {
    int i = blockIdx.x * blockDim.x + threadIdx.x;
    if (i < n) p[i] = v;
}

__global__ void k_fill_u16(unsigned short* p, int n, unsigned short v) {
    int i = blockIdx.x * blockDim.x + threadIdx.x;
    if (i < n) p[i] = v;
}

// Per-column sum and sum of squares. Block handles 128 rows; thread owns 2 cols.
__global__ void k_colstats(const float* __restrict__ x,
                           float* __restrict__ s, float* __restrict__ q, int M) {
    int c0 = threadIdx.x;          // 0..255
    int c1 = threadIdx.x + 256;    // 256..511
    int r0 = blockIdx.x * 128;
    float s0 = 0.f, s1 = 0.f, q0 = 0.f, q1 = 0.f;
    for (int r = 0; r < 128; ++r) {
        int row = r0 + r;
        if (row >= M) break;
        float v0 = x[(size_t)row * DIM + c0];
        float v1 = x[(size_t)row * DIM + c1];
        s0 += v0; q0 += v0 * v0;
        s1 += v1; q1 += v1 * v1;
    }
    atomicAdd(&s[c0], s0); atomicAdd(&s[c1], s1);
    atomicAdd(&q[c0], q0); atomicAdd(&q[c1], q1);
}

__global__ void k_stats_fin(const float* __restrict__ s, const float* __restrict__ q,
                            float* __restrict__ mu, float* __restrict__ inv, int M) {
    int c = blockIdx.x * blockDim.x + threadIdx.x;
    if (c >= DIM) return;
    float m   = s[c] / (float)M;
    float var = q[c] / (float)M - m * m;
    float sd  = sqrtf(fmaxf(var, 0.f));
    mu[c]  = m;
    inv[c] = (sd > 0.f) ? 1.f / sd : 1.f;   // StandardScaler: divide by std where >0 else 1
}

__global__ void k_normalize_bf16(const float* __restrict__ x,
                                 const float* __restrict__ mu, const float* __restrict__ inv,
                                 __bf16* __restrict__ out, int n) {
    int i = blockIdx.x * blockDim.x + threadIdx.x;
    if (i < n) {
        int c = i & (DIM - 1);
        out[i] = (__bf16)((x[i] - mu[c]) * inv[c]);
    }
}

// deg[dst] += 1 for every edge (deg pre-initialized to 1.0 == self loop)
__global__ void k_deg(const long long* __restrict__ ei, float* __restrict__ deg, int E) {
    int e = blockIdx.x * blockDim.x + threadIdx.x;
    if (e < E) atomicAdd(&deg[(int)ei[(size_t)E + e]], 1.0f);
}

__global__ void k_rsqrt_inplace(float* __restrict__ d, int n) {
    int i = blockIdx.x * blockDim.x + threadIdx.x;
    if (i < n) d[i] = rsqrtf(d[i]);   // deg >= 1 always (self loop)
}

// Wt[n*512+k] = bf16(W[k*512+n])  (N-major / "B transposed" for WMMA staging)
__global__ void k_weights_t_bf16(const float* __restrict__ W, __bf16* __restrict__ Wt) {
    int i = blockIdx.x * 256 + threadIdx.x;     // 0..262143
    int n = i >> 9, k = i & (DIM - 1);
    Wt[i] = (__bf16)W[(size_t)k * DIM + n];
}

// ---------------------------------------------------------------------------
// WMMA bf16 GEMM:  C[M x 512] = A[Mpad x 512] (bf16, row major, 128-row padded)
//                             @ B[512 x 512] (given as Bt, N-major bf16)
//                             (+ bias per column, if non-null), f32 out.
// Block tile 128(M) x 64(N), BK = 64, double-buffered async-to-LDS staging.
// 8 waves: 4(M) x 2(N), each wave computes 32x32 = 2x2 WMMA 16x16x32 tiles.
// ---------------------------------------------------------------------------
#define LDA 72   // padded LDS strides (bf16 elems); 72*2B = 144B = 9*16B aligned
#define LDB 72

__global__ __launch_bounds__(256)
void k_gemm_bf16(const __bf16* __restrict__ A, const __bf16* __restrict__ Bt,
                 float* __restrict__ C, const float* __restrict__ bias, int M) {
    __shared__ __align__(16) __bf16 As[2 * 128 * LDA];
    __shared__ __align__(16) __bf16 Bs[2 * 64 * LDB];

    const int tid  = threadIdx.x;
    const int m0   = blockIdx.x * 128;
    const int n0   = blockIdx.y * 64;
    const int wave = tid >> 5, lane = tid & 31;
    const int wm   = (wave & 3) * 32;      // wave's M offset inside block tile
    const int wn   = (wave >> 2) * 32;     // wave's N offset inside block tile
    const int half = lane >> 4;            // 0: lanes 0-15, 1: lanes 16-31
    const int l16  = lane & 15;

    // LDS byte offsets (flat-LDS aperture: low 32 bits == workgroup LDS offset)
    const unsigned asBase = (unsigned)(unsigned long long)&As[0];
    const unsigned bsBase = (unsigned)(unsigned long long)&Bs[0];

    // Issue one K-stage of async global->LDS copies (6 x b128 per thread).
    auto stage = [&](int buf, int kc) {
        unsigned aoff = asBase + (unsigned)buf * (128 * LDA * 2);
        unsigned boff = bsBase + (unsigned)buf * (64 * LDB * 2);
#pragma unroll
        for (int i = 0; i < 4; ++i) {                 // A: 128 rows x 64 k
            int chunk = tid + i * 256;                // 0..1023
            int row = chunk >> 3, k8 = chunk & 7;
            async_cp16(aoff + (unsigned)(row * LDA + k8 * 8) * 2,
                       A + (size_t)(m0 + row) * DIM + kc + k8 * 8);
        }
#pragma unroll
        for (int i = 0; i < 2; ++i) {                 // B: 64 n-rows x 64 k
            int chunk = tid + i * 256;                // 0..511
            int nr = chunk >> 3, k8 = chunk & 7;
            async_cp16(boff + (unsigned)(nr * LDB + k8 * 8) * 2,
                       Bt + (size_t)(n0 + nr) * DIM + kc + k8 * 8);
        }
    };

    v8f c[2][2];
    for (int i = 0; i < 2; ++i)
        for (int j = 0; j < 2; ++j)
            c[i][j] = (v8f){0.f, 0.f, 0.f, 0.f, 0.f, 0.f, 0.f, 0.f};

    stage(0, 0);                                      // prologue prefetch
    int buf = 0;
    for (int kci = 0; kci < DIM / 64; ++kci) {
        if (kci + 1 < DIM / 64) {
            stage(buf ^ 1, (kci + 1) * 64);           // prefetch next stage
            // async loads complete in order: <=6 outstanding => current stage landed
            asm volatile("s_wait_asynccnt 0x6" ::: "memory");
        } else {
            asm volatile("s_wait_asynccnt 0x0" ::: "memory");
        }
        __syncthreads();

        const __bf16* Asb = As + buf * (128 * LDA);
        const __bf16* Bsb = Bs + buf * (64 * LDB);
#pragma unroll
        for (int kk = 0; kk < 2; ++kk) {              // two 32-wide K steps
            const int kb = kk * 32;
            v16bf a[2], b[2];
#pragma unroll
            for (int t = 0; t < 2; ++t) {
                // A frag (16-bit A 16x32 layout): lane l16 = row; lanes<16 hold
                // K 0..7 & 16..23, lanes>=16 hold K 8..15 & 24..31.
                int arow = wm + t * 16 + l16;
                union { v16bf v; bf16x8 h[2]; } ua;
                ua.h[0] = *(const bf16x8*)(&Asb[arow * LDA + kb + half * 8]);
                ua.h[1] = *(const bf16x8*)(&Asb[arow * LDA + kb + 16 + half * 8]);
                a[t] = ua.v;
                // B frag (32x16): lane holds column n = l16; lanes<16 K=0..7 & 16..23,
                // lanes>=16 K=8..15 & 24..31 (same K order as A).
                int brow = wn + t * 16 + l16;
                union { v16bf v; bf16x8 h[2]; } ub;
                ub.h[0] = *(const bf16x8*)(&Bsb[brow * LDB + kb + half * 8]);
                ub.h[1] = *(const bf16x8*)(&Bsb[brow * LDB + kb + 16 + half * 8]);
                b[t] = ub.v;
            }
#pragma unroll
            for (int i = 0; i < 2; ++i)
#pragma unroll
                for (int j = 0; j < 2; ++j)
                    c[i][j] = __builtin_amdgcn_wmma_f32_16x16x32_bf16(
                        false, a[i], false, b[j], (short)0, c[i][j], false, false);
        }
        __syncthreads();
        buf ^= 1;
    }

    // f32 C/D layout: VGPR r -> M = r (lanes 0-15) / 8+r (lanes 16-31); N = l16.
#pragma unroll
    for (int i = 0; i < 2; ++i) {
#pragma unroll
        for (int j = 0; j < 2; ++j) {
            int col = n0 + wn + j * 16 + l16;
            float badd = bias ? bias[col] : 0.f;
#pragma unroll
            for (int r = 0; r < 8; ++r) {
                int row = m0 + wm + i * 16 + half * 8 + r;
                if (row < M) C[(size_t)row * DIM + col] = c[i][j][r] + badd;
            }
        }
    }
}

// ---------------------------------------------------------------------------
// GCN aggregation
// ---------------------------------------------------------------------------
// agg[i][:] = h[i][:] * dis[i]^2   (self-loop term; also initializes agg)
__global__ void k_selfloop_init(const float* __restrict__ h, const float* __restrict__ dis,
                                float* __restrict__ agg, int M) {
    int i = blockIdx.x * blockDim.x + threadIdx.x;   // over M*128 float4s
    int n4 = M * (DIM / 4);
    if (i < n4) {
        int node = i >> 7;
        float d = dis[node];
        float s = d * d;
        float4 v = ((const float4*)h)[i];
        float4 o; o.x = v.x * s; o.y = v.y * s; o.z = v.z * s; o.w = v.w * s;
        ((float4*)agg)[i] = o;
    }
}

// One wave32 per edge: agg[dst] += h[src] * dis[src]*dis[dst]
__global__ void k_edge_agg(const long long* __restrict__ ei, const float* __restrict__ dis,
                           const float* __restrict__ h, float* __restrict__ agg, int E) {
    int gw   = (blockIdx.x * blockDim.x + threadIdx.x) >> 5;
    int lane = threadIdx.x & 31;
    if (gw >= E) return;
    int s = (int)ei[gw];
    int d = (int)ei[(size_t)E + gw];
    float nrm = dis[s] * dis[d];
    const float4* hs = (const float4*)(h + (size_t)s * DIM);
    float* ad = agg + (size_t)d * DIM;
#pragma unroll
    for (int it = 0; it < 4; ++it) {
        int c4 = lane + it * 32;
        float4 v = hs[c4];
        int c = c4 * 4;
        atomicAdd(&ad[c + 0], v.x * nrm);
        atomicAdd(&ad[c + 1], v.y * nrm);
        atomicAdd(&ad[c + 2], v.z * nrm);
        atomicAdd(&ad[c + 3], v.w * nrm);
    }
}

// out_bf16 = bf16(relu(agg + bias[col]))
__global__ void k_bias_relu_bf16(const float* __restrict__ agg, const float* __restrict__ b,
                                 __bf16* __restrict__ out, int n) {
    int i = blockIdx.x * blockDim.x + threadIdx.x;
    if (i < n) {
        int c = i & (DIM - 1);
        out[i] = (__bf16)fmaxf(agg[i] + b[c], 0.f);
    }
}

// One wave32 per node: pooled[batch[node]] += x2[node][:]; cnt[g] += 1
__global__ void k_pool(const long long* __restrict__ batch, const float* __restrict__ x2,
                       float* __restrict__ pooled, float* __restrict__ cnt, int M) {
    int gw   = (blockIdx.x * blockDim.x + threadIdx.x) >> 5;
    int lane = threadIdx.x & 31;
    if (gw >= M) return;
    int g = (int)batch[gw];
    const float4* r = (const float4*)(x2 + (size_t)gw * DIM);
    float* p = pooled + (size_t)g * DIM;
#pragma unroll
    for (int it = 0; it < 4; ++it) {
        int c4 = lane + it * 32;
        float4 v = r[c4];
        int c = c4 * 4;
        atomicAdd(&p[c + 0], v.x);
        atomicAdd(&p[c + 1], v.y);
        atomicAdd(&p[c + 2], v.z);
        atomicAdd(&p[c + 3], v.w);
    }
    if (lane == 0) atomicAdd(&cnt[g], 1.f);
}

// pooled_bf16 = bf16(pooled / max(cnt,1) + b2[col])   (b2 commutes with mean)
__global__ void k_pool_fin_bf16(const float* __restrict__ pooled, const float* __restrict__ cnt,
                                const float* __restrict__ b2, __bf16* __restrict__ out) {
    int i = blockIdx.x * 256 + threadIdx.x;   // N_GRAPHS*DIM = 65536
    int g = i >> 9, c = i & (DIM - 1);
    out[i] = (__bf16)(pooled[i] / fmaxf(cnt[g], 1.f) + b2[c]);
}

// ---------------------------------------------------------------------------
// Host launcher
// ---------------------------------------------------------------------------
extern "C" void kernel_launch(void* const* d_in, const int* in_sizes, int n_in,
                              void* d_out, int out_size, void* d_ws, size_t ws_size,
                              hipStream_t stream) {
    const float*     x     = (const float*)d_in[0];
    const long long* ei    = (const long long*)d_in[1];   // [2, E] int64
    const long long* batch = (const long long*)d_in[2];   // [N] int64
    const float*     W1    = (const float*)d_in[3];
    const float*     b1    = (const float*)d_in[4];
    const float*     W2    = (const float*)d_in[5];
    const float*     b2    = (const float*)d_in[6];
    const float*     Wl    = (const float*)d_in[7];
    const float*     bl    = (const float*)d_in[8];
    float*           out   = (float*)d_out;

    // workspace carve-up (256B aligned)
    char* wsp = (char*)d_ws;
    auto carve = [&](size_t bytes) -> char* {
        char* p = wsp;
        wsp += (bytes + 255) & ~(size_t)255;
        return p;
    };
    float*  mu       = (float*)carve(DIM * 4);
    float*  inv      = (float*)carve(DIM * 4);
    float*  colsum   = (float*)carve(DIM * 4);
    float*  colsumsq = (float*)carve(DIM * 4);
    float*  dis      = (float*)carve((size_t)N_NODES * 4);          // deg -> rsqrt in place
    float*  cnt      = (float*)carve(N_GRAPHS * 4);
    float*  pooled   = (float*)carve((size_t)N_GRAPHS * DIM * 4);
    __bf16* pooledbf = (__bf16*)carve((size_t)N_GRAPHS * DIM * 2);
    __bf16* W1t      = (__bf16*)carve((size_t)DIM * DIM * 2);
    __bf16* W2t      = (__bf16*)carve((size_t)DIM * DIM * 2);
    __bf16* Wlt      = (__bf16*)carve((size_t)DIM * DIM * 2);
    __bf16* xbf      = (__bf16*)carve((size_t)MPAD * DIM * 2);      // xs, then x1 (reused)
    float*  h        = (float*)carve((size_t)N_NODES * DIM * 4);    // GEMM output
    float*  agg      = (float*)carve((size_t)N_NODES * DIM * 4);    // aggregation target

    const int nfeat   = N_NODES * DIM;                 // 10,240,000
    const int npadtail = (MPAD - N_NODES) * DIM;       // 49,152
    const int mblocks = MPAD / 128;                    // 157

    // 1) init
    k_fill_f32<<<4, 256, 0, stream>>>(colsum, DIM, 0.f);
    k_fill_f32<<<4, 256, 0, stream>>>(colsumsq, DIM, 0.f);
    k_fill_f32<<<(N_NODES + 255) / 256, 256, 0, stream>>>(dis, N_NODES, 1.0f); // self loop
    k_fill_f32<<<(N_GRAPHS * DIM + 255) / 256, 256, 0, stream>>>(pooled, N_GRAPHS * DIM, 0.f);
    k_fill_f32<<<1, 256, 0, stream>>>(cnt, N_GRAPHS, 0.f);
    k_fill_u16<<<(npadtail + 255) / 256, 256, 0, stream>>>(
        (unsigned short*)(xbf + (size_t)N_NODES * DIM), npadtail, 0);          // zero pad rows

    // 2) StandardScaler
    k_colstats<<<(N_NODES + 127) / 128, 256, 0, stream>>>(x, colsum, colsumsq, N_NODES);
    k_stats_fin<<<2, 256, 0, stream>>>(colsum, colsumsq, mu, inv, N_NODES);
    k_normalize_bf16<<<(nfeat + 255) / 256, 256, 0, stream>>>(x, mu, inv, xbf, nfeat);

    // 3) degrees -> dis = rsqrt(1 + indeg)
    k_deg<<<(N_EDGES + 255) / 256, 256, 0, stream>>>(ei, dis, N_EDGES);
    k_rsqrt_inplace<<<(N_NODES + 255) / 256, 256, 0, stream>>>(dis, N_NODES);

    // 4) weights -> bf16, N-major
    k_weights_t_bf16<<<DIM * DIM / 256, 256, 0, stream>>>(W1, W1t);
    k_weights_t_bf16<<<DIM * DIM / 256, 256, 0, stream>>>(W2, W2t);
    k_weights_t_bf16<<<DIM * DIM / 256, 256, 0, stream>>>(Wl, Wlt);

    // 5) conv1: h = xs @ W1 ; aggregate ; x1 = bf16(relu(agg + b1))
    k_gemm_bf16<<<dim3(mblocks, DIM / 64), 256, 0, stream>>>(xbf, W1t, h, nullptr, N_NODES);
    k_selfloop_init<<<(N_NODES * (DIM / 4) + 255) / 256, 256, 0, stream>>>(h, dis, agg, N_NODES);
    k_edge_agg<<<(N_EDGES * 32 + 255) / 256, 256, 0, stream>>>(ei, dis, h, agg, N_EDGES);
    k_bias_relu_bf16<<<(nfeat + 255) / 256, 256, 0, stream>>>(agg, b1, xbf, nfeat);

    // 6) conv2: h = x1 @ W2 ; aggregate (bias b2 folded into pooling finalize)
    k_gemm_bf16<<<dim3(mblocks, DIM / 64), 256, 0, stream>>>(xbf, W2t, h, nullptr, N_NODES);
    k_selfloop_init<<<(N_NODES * (DIM / 4) + 255) / 256, 256, 0, stream>>>(h, dis, agg, N_NODES);
    k_edge_agg<<<(N_EDGES * 32 + 255) / 256, 256, 0, stream>>>(ei, dis, h, agg, N_EDGES);

    // 7) global mean pool (+b2), then final linear with fused blin -> d_out
    k_pool<<<(N_NODES * 32 + 255) / 256, 256, 0, stream>>>(batch, agg, pooled, cnt, N_NODES);
    k_pool_fin_bf16<<<N_GRAPHS * DIM / 256, 256, 0, stream>>>(pooled, cnt, b2, pooledbf);
    k_gemm_bf16<<<dim3(1, DIM / 64), 256, 0, stream>>>(pooledbf, Wlt, out, bl, N_GRAPHS);
}
